// vq_ste_56083682951240
// MI455X (gfx1250) — matrix operand order repaired and checked by reading, not compiled, not run on hardware
//
#include <hip/hip_runtime.h>

typedef __attribute__((ext_vector_type(16))) _Float16 v16h;
typedef __attribute__((ext_vector_type(8)))  float    v8f;

#define KCODES 1024
#define CDIM   64
#define NTOT   131072   // 32*64*64 query vectors
#define HWSZ   4096     // H*W

// ---------------------------------------------------------------------------
// Kernel 1: split codebook into f16 hi/lo planes (exact fp32 residual split)
// and precompute ||e_k||^2 in fp32.
// ---------------------------------------------------------------------------
__global__ void vq_prep_kernel(const float* __restrict__ emb,
                               _Float16* __restrict__ ehi,
                               _Float16* __restrict__ elo,
                               float* __restrict__ norms) {
  int k = blockIdx.x * blockDim.x + threadIdx.x;   // 0..1023
  if (k >= KCODES) return;
  const float* row = emb + (size_t)k * CDIM;
  float s = 0.f;
  #pragma unroll 4
  for (int c = 0; c < CDIM; ++c) {
    float v = row[c];
    s = fmaf(v, v, s);
    _Float16 h = (_Float16)v;
    float r = v - (float)h;        // exact in fp32
    _Float16 l = (_Float16)r;
    ehi[(size_t)k * CDIM + c] = h;
    elo[(size_t)k * CDIM + c] = l;
  }
  norms[k] = s;
}

// ---------------------------------------------------------------------------
// Kernel 2: per-wave 32-row tile (two 16x16 WMMA row-tiles sharing B loads).
// acc is seeded with -0.5*||e||^2 so argmin(distance) == argmax(acc).
// Strict-> comparisons keep first-occurrence semantics per lane; exact ties
// across lanes are resolved in the final shuffle reduction.
// ---------------------------------------------------------------------------
__global__ __launch_bounds__(256) void vq_main_kernel(
    const float* __restrict__ z_e,
    const float* __restrict__ emb,
    const _Float16* __restrict__ ehi,
    const _Float16* __restrict__ elo,
    const float* __restrict__ norms,
    float* __restrict__ out) {

  __shared__ float xtile[8][32][65];   // padded stride: conflict-free banks
  __shared__ int   sidx[8][32];

  const int lane = threadIdx.x & 31;
  const int wv   = threadIdx.x >> 5;
  const int tile = blockIdx.x * 8 + wv;        // 0..4095, exact coverage
  const int n0   = tile * 32;
  const int b    = n0 >> 12;                   // n0 / 4096
  const int hw0  = n0 & 4095;

  const int m     = lane & 15;
  const int chalf = lane >> 4;                 // 0 or 1

  // --- stage 32x64 fp32 x-tile into LDS, coalesced per half-wave ----------
  #pragma unroll
  for (int mh = 0; mh < 2; ++mh) {
    #pragma unroll 8
    for (int it = 0; it < 32; ++it) {
      int c = it * 2 + chalf;
      xtile[wv][mh * 16 + m][c] =
          z_e[(((size_t)b * CDIM + c) << 12) + hw0 + mh * 16 + m];
    }
  }
  __builtin_amdgcn_wave_barrier();             // intra-wave LDS RAW ordering

  // --- build A operands in contiguous k order (k = 16*half + s) -------------
  // lane<16: row M=lane, K = cc*32+0..15
  // lane>=16: row M=lane-16, K = cc*32+16..31
  v16h axh[2][2], axl[2][2];                   // [row-tile][c-chunk]
  {
    const int cbase = (lane < 16) ? 0 : 16;
    #pragma unroll
    for (int rt = 0; rt < 2; ++rt) {
      #pragma unroll
      for (int cc = 0; cc < 2; ++cc) {
        #pragma unroll
        for (int j = 0; j < 8; ++j) {
          float v0 = xtile[wv][rt * 16 + m][cc * 32 + cbase + j];
          _Float16 h0 = (_Float16)v0;
          axh[rt][cc][j]     = h0;
          axl[rt][cc][j]     = (_Float16)(v0 - (float)h0);
          float v1 = xtile[wv][rt * 16 + m][cc * 32 + cbase + 8 + j];
          _Float16 h1 = (_Float16)v1;
          axh[rt][cc][8 + j] = h1;
          axl[rt][cc][8 + j] = (_Float16)(v1 - (float)h1);
        }
      }
    }
  }

  float maxv[2][8];
  int   maxi[2][8];
  #pragma unroll
  for (int rt = 0; rt < 2; ++rt)
    #pragma unroll
    for (int v = 0; v < 8; ++v) { maxv[rt][v] = -3.4e38f; maxi[rt][v] = 0; }

  const int col  = lane & 15;                  // C/D column N = lane%16
  const int bofs = (lane >> 4) * 16;           // B: half-wave K split

  // --- sweep all 64 k-tiles of the codebook --------------------------------
  for (int kt = 0; kt < KCODES / 16; ++kt) {
    const int code = kt * 16 + col;
    const size_t rb = (size_t)code * CDIM;
    const v16h bh0 = *(const v16h*)(ehi + rb + bofs);        // c 0..31 chunk
    const v16h bh1 = *(const v16h*)(ehi + rb + 32 + bofs);   // c 32..63 chunk
    const v16h bl0 = *(const v16h*)(elo + rb + bofs);
    const v16h bl1 = *(const v16h*)(elo + rb + 32 + bofs);
    const float s = -0.5f * norms[code];

    v8f acc0 = {s, s, s, s, s, s, s, s};       // acc = dot - 0.5*||e||^2
    v8f acc1 = {s, s, s, s, s, s, s, s};
    acc0 = __builtin_amdgcn_wmma_f32_16x16x32_f16(false, axh[0][0], false, bh0, (short)0, acc0, false, false);
    acc1 = __builtin_amdgcn_wmma_f32_16x16x32_f16(false, axh[1][0], false, bh0, (short)0, acc1, false, false);
    acc0 = __builtin_amdgcn_wmma_f32_16x16x32_f16(false, axh[0][1], false, bh1, (short)0, acc0, false, false);
    acc1 = __builtin_amdgcn_wmma_f32_16x16x32_f16(false, axh[1][1], false, bh1, (short)0, acc1, false, false);
    acc0 = __builtin_amdgcn_wmma_f32_16x16x32_f16(false, axl[0][0], false, bh0, (short)0, acc0, false, false);
    acc1 = __builtin_amdgcn_wmma_f32_16x16x32_f16(false, axl[1][0], false, bh0, (short)0, acc1, false, false);
    acc0 = __builtin_amdgcn_wmma_f32_16x16x32_f16(false, axl[0][1], false, bh1, (short)0, acc0, false, false);
    acc1 = __builtin_amdgcn_wmma_f32_16x16x32_f16(false, axl[1][1], false, bh1, (short)0, acc1, false, false);
    acc0 = __builtin_amdgcn_wmma_f32_16x16x32_f16(false, axh[0][0], false, bl0, (short)0, acc0, false, false);
    acc1 = __builtin_amdgcn_wmma_f32_16x16x32_f16(false, axh[1][0], false, bl0, (short)0, acc1, false, false);
    acc0 = __builtin_amdgcn_wmma_f32_16x16x32_f16(false, axh[0][1], false, bl1, (short)0, acc0, false, false);
    acc1 = __builtin_amdgcn_wmma_f32_16x16x32_f16(false, axh[1][1], false, bl1, (short)0, acc1, false, false);

    // strict > keeps earliest code on exact ties within a lane
    #pragma unroll
    for (int v = 0; v < 8; ++v) {
      bool b0 = acc0[v] > maxv[0][v];
      maxv[0][v] = b0 ? acc0[v] : maxv[0][v];
      maxi[0][v] = b0 ? code    : maxi[0][v];
      bool b1 = acc1[v] > maxv[1][v];
      maxv[1][v] = b1 ? acc1[v] : maxv[1][v];
      maxi[1][v] = b1 ? code    : maxi[1][v];
    }
  }

  // --- argmax reduce across the 16 lanes sharing each row (tie -> min idx) -
  #pragma unroll
  for (int msk = 1; msk < 16; msk <<= 1) {
    #pragma unroll
    for (int rt = 0; rt < 2; ++rt) {
      #pragma unroll
      for (int v = 0; v < 8; ++v) {
        float ov = __shfl_xor(maxv[rt][v], msk, 32);
        int   oi = __shfl_xor(maxi[rt][v], msk, 32);
        bool better = (ov > maxv[rt][v]) ||
                      (ov == maxv[rt][v] && oi < maxi[rt][v]);
        maxv[rt][v] = better ? ov : maxv[rt][v];
        maxi[rt][v] = better ? oi : maxi[rt][v];
      }
    }
  }

  // rows rt*16 + 0..7 live in lanes 0..15; rows rt*16 + 8..15 in lanes 16..31
  if (lane == 0) {
    #pragma unroll
    for (int v = 0; v < 8; ++v) {
      sidx[wv][v]      = maxi[0][v];
      sidx[wv][16 + v] = maxi[1][v];
    }
  }
  if (lane == 16) {
    #pragma unroll
    for (int v = 0; v < 8; ++v) {
      sidx[wv][8 + v]  = maxi[0][v];
      sidx[wv][24 + v] = maxi[1][v];
    }
  }
  __builtin_amdgcn_wave_barrier();

  // --- exact fp32 gather, write out[c*N + n] (reference's transposed order)
  #pragma unroll
  for (int mh = 0; mh < 2; ++mh) {
    const int k2 = sidx[wv][mh * 16 + m];
    #pragma unroll 4
    for (int it = 0; it < 32; ++it) {
      int c = it * 2 + chalf;
      out[(size_t)c * NTOT + n0 + mh * 16 + m] = emb[(size_t)k2 * CDIM + c];
    }
  }
}

// ---------------------------------------------------------------------------
extern "C" void kernel_launch(void* const* d_in, const int* in_sizes, int n_in,
                              void* d_out, int out_size, void* d_ws, size_t ws_size,
                              hipStream_t stream) {
  const float* z_e = (const float*)d_in[0];
  const float* emb = (const float*)d_in[1];
  // d_in[2] = e_dim scalar (compile-time CDIM=64), unused.
  float* out = (float*)d_out;

  _Float16* ehi = (_Float16*)d_ws;                              // 128 KB
  _Float16* elo = ehi + (size_t)KCODES * CDIM;                  // 128 KB
  float* norms  = (float*)((char*)d_ws +
                           2 * (size_t)KCODES * CDIM * sizeof(_Float16)); // 4 KB

  vq_prep_kernel<<<KCODES / 256, 256, 0, stream>>>(emb, ehi, elo, norms);
  vq_main_kernel<<<(NTOT / 32) / 8, 256, 0, stream>>>(z_e, emb, ehi, elo,
                                                      norms, out);
}
